// DeformConv_17824114278916
// MI455X (gfx1250) — matrix operand order repaired and checked by reading, not compiled, not run on hardware
//
#include <hip/hip_runtime.h>
#include <hip/hip_bf16.h>

// ---------- types for WMMA ----------
typedef __attribute__((ext_vector_type(16))) __bf16 v16bf;
typedef __attribute__((ext_vector_type(8)))  __bf16 v8bf;
typedef __attribute__((ext_vector_type(8)))  float  v8f;

union V16u { v16bf v; v8bf h[2]; };

__device__ __forceinline__ unsigned short f2bfu(float f) {
    unsigned u = __float_as_uint(f);
    unsigned r = u + 0x7FFFu + ((u >> 16) & 1u);   // round-to-nearest-even
    return (unsigned short)(r >> 16);
}

#define HW 64
#define HW2 4096          // 64*64
#define CIN 256
#define COUT 256
#define NOFF 18           // 2*K*K

// ---------------------------------------------------------------------------
// Kernel 0: convert weights to bf16 in WMMA-B-friendly [kk][co][ci] layout
// ---------------------------------------------------------------------------
__global__ __launch_bounds__(256) void convert_weights(
    const float* __restrict__ w_offset, const float* __restrict__ w_conv,
    unsigned short* __restrict__ wOffB, unsigned short* __restrict__ wB) {
    int i = blockIdx.x * 256 + threadIdx.x;
    if (i < 9 * 256 * 256) {
        int kk = i >> 16;
        int r  = i & 65535;
        int co = r >> 8, ci = r & 255;
        wB[i] = f2bfu(w_conv[(co * 256 + ci) * 9 + kk]);
    }
    if (i < 9 * 32 * 256) {
        int kk = i / 8192;
        int r  = i & 8191;
        int co = r >> 8, ci = r & 255;
        float v = (co < NOFF) ? w_offset[(co * 256 + ci) * 9 + kk] : 0.f;
        wOffB[i] = f2bfu(v);
    }
}

// ---------------------------------------------------------------------------
// Kernel 1: offset conv (3x3, 256 -> 18) via bf16 WMMA.
// Block = 64 threads (2 waves). M = 16 x-positions, N = 32 co (2 tiles).
// ---------------------------------------------------------------------------
__global__ __launch_bounds__(64) void offset_conv(
    const float* __restrict__ x, const unsigned short* __restrict__ wOffB,
    const float* __restrict__ b_offset, float* __restrict__ offs) {
    const int xb  = blockIdx.x * 16;
    const int y   = blockIdx.y;
    const int n   = blockIdx.z;
    const int tid = threadIdx.x;
    const int lane = tid & 31;
    const int w    = tid >> 5;
    const int row  = lane & 15;
    const int hi   = lane >> 4;

    __shared__ __align__(32) unsigned short Abuf[16 * 256];

    v8f acc = {};
    const int co0 = w * 16;

    for (int kk = 0; kk < 9; ++kk) {
        const int ky = kk / 3, kx = kk - ky * 3;
        const int ys = y - 1 + ky;
        for (int i = tid; i < 16 * 256; i += 64) {
            int m = i & 15, ci = i >> 4;
            int xs = xb + m - 1 + kx;
            float v = 0.f;
            if (ys >= 0 && ys < HW && xs >= 0 && xs < HW)
                v = x[(size_t)(n * CIN + ci) * HW2 + ys * HW + xs];
            Abuf[m * 256 + ci] = f2bfu(v);
        }
        __syncthreads();

        const unsigned short* aBase = Abuf + row * 256 + hi * 16;
        const unsigned short* bBase = wOffB + ((size_t)(kk * 32 + co0 + row)) * 256 + hi * 16;
#pragma unroll
        for (int c = 0; c < 8; ++c) {
            V16u a;
            a.h[0] = *reinterpret_cast<const v8bf*>(aBase + c * 32);
            a.h[1] = *reinterpret_cast<const v8bf*>(aBase + c * 32 + 8);
            v16bf b = *reinterpret_cast<const v16bf*>(bBase + c * 32);
            acc = __builtin_amdgcn_wmma_f32_16x16x32_bf16(
                false, a.v, false, b, (short)0, acc, false, false);
        }
        __syncthreads();
    }

    const int co = co0 + row;
    if (co < NOFF) {
        float bias = b_offset[co];
        float* op = offs + ((size_t)(n * NOFF + co) * HW + y) * HW + xb;
#pragma unroll
        for (int r = 0; r < 8; ++r) {
            int m = r + (hi << 3);
            op[m] = acc[r] + bias;
        }
    }
}

// ---------------------------------------------------------------------------
// Kernel 2: deformable im2col + GEMM via bf16 WMMA.
// Block = 256 threads (8 waves). Tile: M = 32 x-positions (2 m-tiles),
// N = 256 co (2 x 16-co tiles per wave), K = 2304 (9 taps x 8 chunks of 32).
// Double-buffered LDS im2col panel: sample tap kk+1 while tap kk is consumed.
// ---------------------------------------------------------------------------
__global__ __launch_bounds__(256) void deform_gemm(
    const float* __restrict__ x, const float* __restrict__ offs,
    const unsigned short* __restrict__ wB, float* __restrict__ out_raw) {
    const int xb  = blockIdx.x * 32;
    const int y   = blockIdx.y;
    const int n   = blockIdx.z;
    const int tid = threadIdx.x;
    const int lane = tid & 31;
    const int w    = tid >> 5;            // wave 0..7
    const int row  = lane & 15;
    const int hi   = lane >> 4;

    __shared__ int   sIdx[32 * 9][4];
    __shared__ float sWgt[32 * 9][4];
    __shared__ __align__(32) unsigned short Abuf[2][32 * 256];   // 2 x 16 KB

    // precompute bilinear taps for 32 positions x 9 kernel points
    for (int t = tid; t < 32 * 9; t += 256) {
        int m = t / 9, kk = t - m * 9;
        int xm = xb + m;
        int ky = kk / 3, kx = kk - ky * 3;
        float dy = offs[((size_t)(n * NOFF + 2 * kk) * HW + y) * HW + xm];
        float dx = offs[((size_t)(n * NOFF + 2 * kk + 1) * HW + y) * HW + xm];
        float py = (float)(y - 1 + ky) + dy;
        float px = (float)(xm - 1 + kx) + dx;
        float fy = floorf(py), fx = floorf(px);
        int y0 = (int)fy, x0 = (int)fx;
        float ly = py - fy, lx = px - fx;
        int y1 = y0 + 1, x1 = x0 + 1;
        bool vy0 = (y0 >= 0) & (y0 < HW), vy1 = (y1 >= 0) & (y1 < HW);
        bool vx0 = (x0 >= 0) & (x0 < HW), vx1 = (x1 >= 0) & (x1 < HW);
        int yc0 = min(max(y0, 0), HW - 1), yc1 = min(max(y1, 0), HW - 1);
        int xc0 = min(max(x0, 0), HW - 1), xc1 = min(max(x1, 0), HW - 1);
        sIdx[t][0] = yc0 * HW + xc0;
        sIdx[t][1] = yc0 * HW + xc1;
        sIdx[t][2] = yc1 * HW + xc0;
        sIdx[t][3] = yc1 * HW + xc1;
        sWgt[t][0] = (vy0 && vx0) ? (1.f - ly) * (1.f - lx) : 0.f;
        sWgt[t][1] = (vy0 && vx1) ? (1.f - ly) * lx : 0.f;
        sWgt[t][2] = (vy1 && vx0) ? ly * (1.f - lx) : 0.f;
        sWgt[t][3] = (vy1 && vx1) ? ly * lx : 0.f;
    }
    __syncthreads();

    // Deformable sampling of one tap into one LDS buffer.
    // lane-layout: consecutive tids sample consecutive x-positions (coalesced).
    const int sm = tid & 31;          // x-position owned by this thread
    auto sample = [&](int kk, int buf) {
        const int it = sm * 9 + kk;
        const int i0 = sIdx[it][0], i1 = sIdx[it][1];
        const int i2 = sIdx[it][2], i3 = sIdx[it][3];
        const float w0 = sWgt[it][0], w1 = sWgt[it][1];
        const float w2 = sWgt[it][2], w3 = sWgt[it][3];
        const float* xn = x + (size_t)n * CIN * HW2;
        unsigned short* ab = &Abuf[buf][sm * 256];
        for (int ci = (tid >> 5); ci < 256; ci += 8) {
            const float* xp = xn + (size_t)ci * HW2;
            float v = w0 * xp[i0] + w1 * xp[i1] + w2 * xp[i2] + w3 * xp[i3];
            ab[ci] = f2bfu(v);
        }
    };

    v8f acc00 = {}, acc01 = {};       // m-tile 0 x {coA, coB}
    v8f acc10 = {}, acc11 = {};       // m-tile 1 x {coA, coB}
    const int coA = w * 32;
    const int coB = w * 32 + 16;

    sample(0, 0);
    __syncthreads();

    for (int kk = 0; kk < 9; ++kk) {
        const int cur = kk & 1;
        if (kk < 8) {
            sample(kk + 1, cur ^ 1);
            // prefetch next tap's weight rows into near caches
            __builtin_prefetch(wB + ((size_t)((kk + 1) * 256 + coA + row)) * 256, 0, 3);
            __builtin_prefetch(wB + ((size_t)((kk + 1) * 256 + coB + row)) * 256, 0, 3);
        }

        const unsigned short* aB0 = &Abuf[cur][row * 256] + hi * 16;          // m-tile 0
        const unsigned short* aB1 = &Abuf[cur][(16 + row) * 256] + hi * 16;   // m-tile 1
        const unsigned short* bBaseA = wB + ((size_t)(kk * 256 + coA + row)) * 256 + hi * 16;
        const unsigned short* bBaseB = wB + ((size_t)(kk * 256 + coB + row)) * 256 + hi * 16;
#pragma unroll
        for (int c = 0; c < 8; ++c) {
            V16u a0, a1;
            a0.h[0] = *reinterpret_cast<const v8bf*>(aB0 + c * 32);
            a0.h[1] = *reinterpret_cast<const v8bf*>(aB0 + c * 32 + 8);
            a1.h[0] = *reinterpret_cast<const v8bf*>(aB1 + c * 32);
            a1.h[1] = *reinterpret_cast<const v8bf*>(aB1 + c * 32 + 8);
            v16bf bA = *reinterpret_cast<const v16bf*>(bBaseA + c * 32);
            v16bf bB = *reinterpret_cast<const v16bf*>(bBaseB + c * 32);
            acc00 = __builtin_amdgcn_wmma_f32_16x16x32_bf16(
                false, a0.v, false, bA, (short)0, acc00, false, false);
            acc01 = __builtin_amdgcn_wmma_f32_16x16x32_bf16(
                false, a0.v, false, bB, (short)0, acc01, false, false);
            acc10 = __builtin_amdgcn_wmma_f32_16x16x32_bf16(
                false, a1.v, false, bA, (short)0, acc10, false, false);
            acc11 = __builtin_amdgcn_wmma_f32_16x16x32_bf16(
                false, a1.v, false, bB, (short)0, acc11, false, false);
        }
        __syncthreads();
    }

    // D layout: VGPR r -> (M = r + 8*hi, N = row)
    float* op = out_raw + (size_t)n * COUT * HW2 + (size_t)y * HW + xb;
    const size_t cA = (size_t)(coA + row) * HW2;
    const size_t cB = (size_t)(coB + row) * HW2;
#pragma unroll
    for (int r = 0; r < 8; ++r) {
        int m0 = r + (hi << 3);
        int m1 = m0 + 16;
        op[cA + m0] = acc00[r];
        op[cB + m0] = acc01[r];
        op[cA + m1] = acc10[r];
        op[cB + m1] = acc11[r];
    }
}

// ---------------------------------------------------------------------------
// Kernel 3: per-channel BN statistics -> folded (scale, shift)
// ---------------------------------------------------------------------------
__global__ __launch_bounds__(256) void bn_stats(
    const float* __restrict__ out_raw, const float* __restrict__ gamma,
    const float* __restrict__ beta, float* __restrict__ sc_sh) {
    const int co = blockIdx.x;
    const int tid = threadIdx.x;
    float s = 0.f, s2 = 0.f;
    for (int i = tid; i < 8 * HW2; i += 256) {
        int nn = i >> 12;
        int p  = i & (HW2 - 1);
        float v = out_raw[((size_t)(nn * COUT + co)) * HW2 + p];
        s += v; s2 += v * v;
    }
    __shared__ float rs[256], rs2[256];
    rs[tid] = s; rs2[tid] = s2;
    __syncthreads();
    for (int off = 128; off > 0; off >>= 1) {
        if (tid < off) { rs[tid] += rs[tid + off]; rs2[tid] += rs2[tid + off]; }
        __syncthreads();
    }
    if (tid == 0) {
        const float inv = 1.f / (8.f * (float)HW2);
        float mean = rs[0] * inv;
        float var  = rs2[0] * inv - mean * mean;
        float rstd = rsqrtf(var + 1e-5f);
        float scale = gamma[co] * rstd;
        sc_sh[co]        = scale;
        sc_sh[256 + co]  = beta[co] - mean * scale;
    }
}

// ---------------------------------------------------------------------------
// Kernel 4: normalize + SiLU, vectorized float4
// ---------------------------------------------------------------------------
__global__ __launch_bounds__(256) void bn_silu(
    const float* __restrict__ out_raw, const float* __restrict__ sc_sh,
    float* __restrict__ out) {
    size_t i = (size_t)blockIdx.x * 256 + threadIdx.x;   // float4 index
    const size_t total4 = (size_t)8 * COUT * HW2 / 4;
    if (i >= total4) return;
    float4 v = reinterpret_cast<const float4*>(out_raw)[i];
    int co = (int)((i * 4) >> 12) & 255;
    float sc = sc_sh[co], sh = sc_sh[256 + co];
    float4 o;
    float t;
    t = v.x * sc + sh; o.x = t / (1.f + __expf(-t));
    t = v.y * sc + sh; o.y = t / (1.f + __expf(-t));
    t = v.z * sc + sh; o.z = t / (1.f + __expf(-t));
    t = v.w * sc + sh; o.w = t / (1.f + __expf(-t));
    reinterpret_cast<float4*>(out)[i] = o;
}

// ---------------------------------------------------------------------------
extern "C" void kernel_launch(void* const* d_in, const int* in_sizes, int n_in,
                              void* d_out, int out_size, void* d_ws, size_t ws_size,
                              hipStream_t stream) {
    const float* x        = (const float*)d_in[0];  // [8,256,64,64]
    const float* w_offset = (const float*)d_in[1];  // [18,256,3,3]
    const float* b_offset = (const float*)d_in[2];  // [18]
    const float* w_conv   = (const float*)d_in[3];  // [256,256,3,3]
    const float* gamma    = (const float*)d_in[4];  // [256]
    const float* beta     = (const float*)d_in[5];  // [256]
    float* out = (float*)d_out;

    // workspace carve-up
    float* offs    = (float*)d_ws;                         // 8*18*64*64
    float* out_raw = offs + (size_t)8 * NOFF * HW2;        // 8*256*64*64
    float* sc_sh   = out_raw + (size_t)8 * COUT * HW2;     // 512
    unsigned short* wOffB = (unsigned short*)(sc_sh + 512);   // 9*32*256 bf16
    unsigned short* wB    = wOffB + (size_t)9 * 32 * 256;     // 9*256*256 bf16

    convert_weights<<<2304, 256, 0, stream>>>(w_offset, w_conv, wOffB, wB);

    offset_conv<<<dim3(4, 64, 8), 64, 0, stream>>>(x, wOffB, b_offset, offs);
    deform_gemm<<<dim3(2, 64, 8), 256, 0, stream>>>(x, offs, wB, out_raw);

    bn_stats<<<256, 256, 0, stream>>>(out_raw, gamma, beta, sc_sh);
    bn_silu<<<8192, 256, 0, stream>>>(out_raw, sc_sh, out);
}
